// GATEncoder_498216206890
// MI455X (gfx1250) — matrix operand order repaired and checked by reading, not compiled, not run on hardware
//
#include <hip/hip_runtime.h>
#include <math.h>

#define FC 256            // heads*ch == 256 for both layers
#define KIN 20
#define NEG_SLOPE 0.2f
#define SM_EPS 1e-16f

typedef _Float16 v8h  __attribute__((ext_vector_type(8)));
typedef _Float16 v16h __attribute__((ext_vector_type(16)));
typedef float    v8f  __attribute__((ext_vector_type(8)));

// ---------- monotone float<->uint mapping for atomicMax on floats ----------
__device__ __forceinline__ unsigned enc_f32(float f) {
  unsigned u = __float_as_uint(f);
  return (u & 0x80000000u) ? ~u : (u | 0x80000000u);
}
__device__ __forceinline__ float dec_f32(unsigned k) {
  return __uint_as_float((k & 0x80000000u) ? (k ^ 0x80000000u) : ~k);
}

// ---------- init ----------
__global__ void fill_f32(float* __restrict__ p, float v, long n) {
  long i = (long)blockIdx.x * blockDim.x + threadIdx.x;
  if (i < n) p[i] = v;
}
__global__ void fill_u32(unsigned* __restrict__ p, unsigned v, long n) {
  long i = (long)blockIdx.x * blockDim.x + threadIdx.x;
  if (i < n) p[i] = v;
}

// ---------- f32 -> f16 conversions ----------
// src [M][Kin] f32  ->  dst [M][Kpad] f16 (zero padded)
__global__ void pad_convert_a16(const float* __restrict__ src, _Float16* __restrict__ dst,
                                int Kin, int Kpad, long n) {
  long i = (long)blockIdx.x * blockDim.x + threadIdx.x;
  if (i >= n) return;
  long r = i / Kpad;
  int  k = (int)(i % Kpad);
  dst[i] = (k < Kin) ? (_Float16)src[r * Kin + k] : (_Float16)0.0f;
}
// W [Kin][Ncol] f32 -> Bt [Ncol][Kpad] f16 (transposed, zero padded)
__global__ void transpose_convert_b16(const float* __restrict__ W, _Float16* __restrict__ Bt,
                                      int Kin, int Kpad, int Ncol) {
  int i = blockIdx.x * blockDim.x + threadIdx.x;
  if (i >= Ncol * Kpad) return;
  int nn = i / Kpad;
  int k  = i % Kpad;
  Bt[i] = (k < Kin) ? (_Float16)W[(size_t)k * Ncol + nn] : (_Float16)0.0f;
}

// ---------- WMMA GEMM: C[M][256] = A[M][Kpad] * B(Kin x 256) + bias ----------
// Each wave computes a 16x64 output slab: one shared A fragment per K-step feeds
// 4 WMMAs (4 adjacent 16x16 column tiles). Block = 8 waves = 2 row tiles x 16 col tiles.
// A fragment: 16-bit A 16x32 ISA layout (lane m=L%16; K = kb..kb+7 and 16+kb..16+kb+7, kb=(L<16)?0:8)
// B fragment: same K order as A (lane n=L%16; K = kb..kb+7 and 16+kb..16+kb+7, kb=(L<16)?0:8)
//   -> Bt stored [n][k] so each B fragment is two contiguous 16B loads.
template <int KPAD>
__global__ void __launch_bounds__(256)
gemm_wmma_f16(const _Float16* __restrict__ A,   // [M][KPAD]
              const _Float16* __restrict__ Bt,  // [256][KPAD]
              const float* __restrict__ bias,   // [256]
              float* __restrict__ C,            // [M][256]
              int M) {
  const int lane = threadIdx.x & 31;
  const int wave = threadIdx.x >> 5;
  const int rowT = blockIdx.x * 2 + (wave >> 2);  // 2 row tiles per block
  const int cg   = wave & 3;                      // column group: 4 tiles of 16
  const int hi   = lane >> 4;
  const int l16  = lane & 15;
  int m = rowT * 16 + l16; if (m >= M) m = M - 1; // clamp loads; stores guarded below
  const _Float16* ap = A  + (size_t)m * KPAD + hi * 8;
  const _Float16* bp = Bt + (size_t)(cg * 64 + l16) * KPAD + hi * 8;

  v8f acc[4];
#pragma unroll
  for (int c = 0; c < 4; ++c) acc[c] = (v8f){0.f, 0.f, 0.f, 0.f, 0.f, 0.f, 0.f, 0.f};

#pragma unroll
  for (int k0 = 0; k0 < KPAD; k0 += 32) {
    v8h alo = *(const v8h*)(ap + k0);
    v8h ahi = *(const v8h*)(ap + k0 + 16);
    v16h a;
#pragma unroll
    for (int i = 0; i < 8; ++i) { a[i] = alo[i]; a[i + 8] = ahi[i]; }
#pragma unroll
    for (int c = 0; c < 4; ++c) {
      v8h blo = *(const v8h*)(bp + (size_t)c * 16 * KPAD + k0);
      v8h bhi = *(const v8h*)(bp + (size_t)c * 16 * KPAD + k0 + 16);
      v16h b;
#pragma unroll
      for (int i = 0; i < 8; ++i) { b[i] = blo[i]; b[i + 8] = bhi[i]; }
      acc[c] = __builtin_amdgcn_wmma_f32_16x16x32_f16(false, a, false, b, (short)0,
                                                      acc[c], false, false);
    }
  }

  // C layout: lanes 0-15 hold rows rbase..rbase+7, lanes 16-31 rows rbase+8..rbase+15
  const int rbase = rowT * 16 + hi * 8;
  if (rowT * 16 + 16 <= M) {            // uniform fast path: full tile, no divergence
#pragma unroll
    for (int c = 0; c < 4; ++c) {
      const int n = cg * 64 + c * 16 + l16;
      const float bn = bias[n];
#pragma unroll
      for (int r = 0; r < 8; ++r) C[(size_t)(rbase + r) * FC + n] = acc[c][r] + bn;
    }
  } else {
#pragma unroll
    for (int c = 0; c < 4; ++c) {
      const int n = cg * 64 + c * 16 + l16;
      const float bn = bias[n];
#pragma unroll
      for (int r = 0; r < 8; ++r)
        if (rbase + r < M) C[(size_t)(rbase + r) * FC + n] = acc[c][r] + bn;
    }
  }
}

// ---------- edge kernels (one wave32 per edge; lane covers 8 channels) ----------
template <int HEADS>
__global__ void __launch_bounds__(256)
edge_logits(const float* __restrict__ xl, const float* __restrict__ xr,
            const float* __restrict__ att, const int* __restrict__ ei,
            float* __restrict__ logits, unsigned* __restrict__ maxkey,
            int E, int Etot) {
  const int lane = threadIdx.x & 31;
  const int e = blockIdx.x * 8 + (threadIdx.x >> 5);
  if (e >= Etot) return;
  int src, dst;
  if (e < E) { src = ei[e]; dst = ei[E + e]; } else { src = dst = e - E; }
  const int base = lane * 8;
  const float4* pa = (const float4*)(xl + (size_t)src * FC + base);
  const float4* pb = (const float4*)(xr + (size_t)dst * FC + base);
  const float4* pt = (const float4*)(att + base);
  float acc = 0.f;
#pragma unroll
  for (int q = 0; q < 2; ++q) {
    float4 a = pa[q], b = pb[q], t = pt[q];
    float s;
    s = a.x + b.x; acc += t.x * (s > 0.f ? s : NEG_SLOPE * s);
    s = a.y + b.y; acc += t.y * (s > 0.f ? s : NEG_SLOPE * s);
    s = a.z + b.z; acc += t.z * (s > 0.f ? s : NEG_SLOPE * s);
    s = a.w + b.w; acc += t.w * (s > 0.f ? s : NEG_SLOPE * s);
  }
  constexpr int LPH = 32 / HEADS;               // lanes per head
#pragma unroll
  for (int off = 1; off < LPH; off <<= 1) acc += __shfl_xor(acc, off, 32);
  if ((lane & (LPH - 1)) == 0) {
    const int h = lane / LPH;
    logits[(size_t)e * HEADS + h] = acc;
    atomicMax(maxkey + (size_t)dst * HEADS + h, enc_f32(acc));
  }
}

template <int HEADS>
__global__ void edge_expw(const int* __restrict__ ei, float* __restrict__ logits,
                          const unsigned* __restrict__ maxkey, float* __restrict__ sums,
                          int E, int Etot) {
  long idx = (long)blockIdx.x * blockDim.x + threadIdx.x;
  if (idx >= (long)Etot * HEADS) return;
  int e = (int)(idx / HEADS);
  int h = (int)(idx % HEADS);
  int dst = (e < E) ? ei[E + e] : e - E;
  float m = dec_f32(maxkey[(size_t)dst * HEADS + h]);
  float w = __expf(logits[idx] - m);
  logits[idx] = w;
  atomicAdd(sums + (size_t)dst * HEADS + h, w);
}

template <int HEADS>
__global__ void __launch_bounds__(256)
edge_aggregate(const float* __restrict__ xl, const float* __restrict__ logits,
               const float* __restrict__ sums, const int* __restrict__ ei,
               float* __restrict__ out, int E, int Etot) {
  const int lane = threadIdx.x & 31;
  const int e = blockIdx.x * 8 + (threadIdx.x >> 5);
  if (e >= Etot) return;
  int src, dst;
  if (e < E) { src = ei[e]; dst = ei[E + e]; } else { src = dst = e - E; }
  constexpr int LPH = 32 / HEADS;
  const int h = lane / LPH;
  float w = logits[(size_t)e * HEADS + h];
  float s = sums[(size_t)dst * HEADS + h];
  float alpha = w / (s + SM_EPS);
  const int base = lane * 8;
  const float* ps = xl + (size_t)src * FC + base;
  float* pd = out + (size_t)dst * FC + base;
#pragma unroll
  for (int k = 0; k < 8; ++k) atomicAdd(pd + k, alpha * ps[k]);
}

// ---------- pointwise ----------
__global__ void elu_bias(float* __restrict__ h, const float* __restrict__ bias, long n) {
  long i = (long)blockIdx.x * blockDim.x + threadIdx.x;
  if (i >= n) return;
  float v = h[i] + bias[i % FC];
  h[i] = v > 0.f ? v : (__expf(v) - 1.f);
}
__global__ void add_bias(float* __restrict__ o, const float* __restrict__ bias, long n) {
  long i = (long)blockIdx.x * blockDim.x + threadIdx.x;
  if (i >= n) return;
  o[i] += bias[i % FC];
}

static inline size_t al256(size_t o) { return (o + 255) & ~(size_t)255; }

extern "C" void kernel_launch(void* const* d_in, const int* in_sizes, int n_in,
                              void* d_out, int out_size, void* d_ws, size_t ws_size,
                              hipStream_t stream) {
  const float* x     = (const float*)d_in[0];
  const int*   ei    = (const int*)d_in[1];
  const float* W_l1  = (const float*)d_in[2];
  const float* b_l1  = (const float*)d_in[3];
  const float* W_r1  = (const float*)d_in[4];
  const float* b_r1  = (const float*)d_in[5];
  const float* att1  = (const float*)d_in[6];
  const float* bias1 = (const float*)d_in[7];
  const float* W_l2  = (const float*)d_in[8];
  const float* b_l2  = (const float*)d_in[9];
  const float* W_r2  = (const float*)d_in[10];
  const float* b_r2  = (const float*)d_in[11];
  const float* att2  = (const float*)d_in[12];
  const float* bias2 = (const float*)d_in[13];

  const int N    = in_sizes[0] / KIN;
  const int E    = in_sizes[1] / 2;
  const int Etot = E + N;
  float* out = (float*)d_out;   // also used as scratch for h between layers

  // workspace carve
  char* w = (char*)d_ws;
  size_t o = 0;
  float*     xl     = (float*)(w + o);     o = al256(o + (size_t)N * FC * 4);
  float*     xr     = (float*)(w + o);     o = al256(o + (size_t)N * FC * 4);
  float*     logits = (float*)(w + o);     o = al256(o + (size_t)Etot * 4 * 4);
  unsigned*  maxkey = (unsigned*)(w + o);  o = al256(o + (size_t)N * 4 * 4);
  float*     sums   = (float*)(w + o);     o = al256(o + (size_t)N * 4 * 4);
  _Float16*  a16    = (_Float16*)(w + o);  o = al256(o + (size_t)N * FC * 2);
  _Float16*  bt16a  = (_Float16*)(w + o);  o = al256(o + (size_t)FC * FC * 2);
  _Float16*  bt16b  = (_Float16*)(w + o);  o = al256(o + (size_t)FC * FC * 2);

  const int T = 256;
  const int mTiles = (N + 15) / 16;
  const dim3 gGemm((mTiles + 1) / 2);                // 8 waves: 2 row tiles x 16 col tiles
  const int eBlocks = (Etot + 7) / 8;                // wave-per-edge kernels
  const long nNF = (long)N * FC;
  auto blks = [](long n, int t) { return (unsigned)((n + t - 1) / t); };

  // ===================== Layer 1 (heads=4, ch=64) =====================
  fill_u32<<<blks((long)N * 4, T), T, 0, stream>>>(maxkey, 0u, (long)N * 4);
  fill_f32<<<blks((long)N * 4, T), T, 0, stream>>>(sums, 0.f, (long)N * 4);

  pad_convert_a16<<<blks((long)N * 32, T), T, 0, stream>>>(x, a16, KIN, 32, (long)N * 32);
  transpose_convert_b16<<<blks(FC * 32, T), T, 0, stream>>>(W_l1, bt16a, KIN, 32, FC);
  transpose_convert_b16<<<blks(FC * 32, T), T, 0, stream>>>(W_r1, bt16b, KIN, 32, FC);
  gemm_wmma_f16<32><<<gGemm, T, 0, stream>>>(a16, bt16a, b_l1, xl, N);
  gemm_wmma_f16<32><<<gGemm, T, 0, stream>>>(a16, bt16b, b_r1, xr, N);

  edge_logits<4><<<eBlocks, T, 0, stream>>>(xl, xr, att1, ei, logits, maxkey, E, Etot);
  edge_expw<4><<<blks((long)Etot * 4, T), T, 0, stream>>>(ei, logits, maxkey, sums, E, Etot);
  fill_f32<<<blks(nNF, T), T, 0, stream>>>(out, 0.f, nNF);       // h accumulator
  edge_aggregate<4><<<eBlocks, T, 0, stream>>>(xl, logits, sums, ei, out, E, Etot);
  elu_bias<<<blks(nNF, T), T, 0, stream>>>(out, bias1, nNF);     // h = elu(agg + bias1)

  // ===================== Layer 2 (heads=1, ch=256) =====================
  pad_convert_a16<<<blks(nNF, T), T, 0, stream>>>(out, a16, FC, FC, nNF);  // h -> f16
  transpose_convert_b16<<<blks(FC * FC, T), T, 0, stream>>>(W_l2, bt16a, FC, FC, FC);
  transpose_convert_b16<<<blks(FC * FC, T), T, 0, stream>>>(W_r2, bt16b, FC, FC, FC);
  gemm_wmma_f16<256><<<gGemm, T, 0, stream>>>(a16, bt16a, b_l2, xl, N);
  gemm_wmma_f16<256><<<gGemm, T, 0, stream>>>(a16, bt16b, b_r2, xr, N);

  fill_u32<<<blks((long)N, T), T, 0, stream>>>(maxkey, 0u, (long)N);
  fill_f32<<<blks((long)N, T), T, 0, stream>>>(sums, 0.f, (long)N);
  edge_logits<1><<<eBlocks, T, 0, stream>>>(xl, xr, att2, ei, logits, maxkey, E, Etot);
  edge_expw<1><<<blks((long)Etot, T), T, 0, stream>>>(ei, logits, maxkey, sums, E, Etot);
  fill_f32<<<blks(nNF, T), T, 0, stream>>>(out, 0.f, nNF);       // final accumulator
  edge_aggregate<1><<<eBlocks, T, 0, stream>>>(xl, logits, sums, ei, out, E, Etot);
  add_bias<<<blks(nNF, T), T, 0, stream>>>(out, bias2, nNF);
}
